// SparseBlock_67705864454829
// MI455X (gfx1250) — matrix operand order repaired and checked by reading, not compiled, not run on hardware
//
#include <hip/hip_runtime.h>
#include <hip/hip_bf16.h>

typedef __attribute__((ext_vector_type(16))) __bf16 v16bf;
typedef __attribute__((ext_vector_type(8)))  float  v8f;

#define C_DIM 128
#define T_DIM 128
#define F_DIM 128
#define B_DIM 32
#define K_SEL 512
#define NPIX  (T_DIM * F_DIM)   // 16384

// conv tiling
#define TT 4          // t rows per block
#define FT 32         // f cols per block
#define SLAB_R (TT + 2)
#define SLAB_Q (FT + 2)
#define CIP 136       // padded ci stride (272B -> conflict-free, 16B aligned)

// workspace byte offsets
#define WS_WCONV 0ull                      // 9*128*128 bf16 = 294912 B
#define WS_WVAL  294912ull                 // 128*128 bf16   = 32768 B
#define WS_H     327680ull                 // 32*128*16384 bf16 = 134217728 B
#define WS_PROBS 134545408ull              // 32*16384 f32 = 2097152 B
#define WS_SEL   136642560ull              // 32*512 i32 = 65536 B

// ---------------------------------------------------------------------------
// 1) repack weights to bf16; conv weights -> [tap][co][ci]
// ---------------------------------------------------------------------------
__global__ __launch_bounds__(256) void convert_w_kernel(
    const float* __restrict__ conv_w, const float* __restrict__ values_w,
    __bf16* __restrict__ wconv, __bf16* __restrict__ wval) {
  int i = blockIdx.x * 256 + threadIdx.x;
  const int NC = 9 * C_DIM * C_DIM;
  if (i < NC) {
    int tap = i / (C_DIM * C_DIM);
    int co  = (i / C_DIM) % C_DIM;
    int ci  = i % C_DIM;
    wconv[i] = (__bf16)conv_w[(co * C_DIM + ci) * 9 + tap];
  } else if (i < NC + C_DIM * C_DIM) {
    int j = i - NC;
    wval[j] = (__bf16)values_w[j];
  }
}

// ---------------------------------------------------------------------------
// 2) implicit-GEMM conv (bf16 WMMA) fused with channel-norm + select logit
//    block = 256 threads = 8 waves; tile = 4x32 pixels, all 128 C_out.
//    Per-wave: 4 co-tiles x 2 pixel-tiles (cuts LDS B-fragment traffic 4x
//    vs 1x8; A fragments come from the L0/L2-resident 290KB weight block).
// ---------------------------------------------------------------------------
__global__ __launch_bounds__(256) void conv_select_kernel(
    const float* __restrict__ x, const __bf16* __restrict__ wconv,
    const float* __restrict__ conv_b, const float* __restrict__ select_w,
    const float* __restrict__ select_b, __bf16* __restrict__ h_out,
    float* __restrict__ logits) {
  __shared__ __bf16 lds_x[SLAB_R][SLAB_Q][CIP];
  __shared__ float lds_p1[TT * FT][4];
  __shared__ float lds_p2[TT * FT][4];
  __shared__ float lds_sw[C_DIM];
  __shared__ float lds_cb[C_DIM];

  const int tid  = threadIdx.x;
  const int lane = tid & 31;
  const int wv   = tid >> 5;
  const int ln   = lane & 15;
  const int half = lane >> 4;

  const int mgrp = wv & 1;   // co-tiles mgrp*4 .. mgrp*4+3
  const int ngrp = wv >> 1;  // pixel-tiles ngrp*2, ngrp*2+1

  const int f0 = blockIdx.x * FT;
  const int t0 = blockIdx.y * TT;
  const int b  = blockIdx.z;

  if (tid < C_DIM) { lds_sw[tid] = select_w[tid]; lds_cb[tid] = conv_b[tid]; }

  // stage input slab (rows t0-1..t0+4, cols f0-1..f0+32, all ci) zero-padded
  const int SLAB_ELEMS = SLAB_R * SLAB_Q * C_DIM;
  for (int i = tid; i < SLAB_ELEMS; i += 256) {
    int col = i % SLAB_Q;
    int tmp = i / SLAB_Q;
    int ci  = tmp % C_DIM;
    int r   = tmp / C_DIM;
    int t = t0 + r - 1;
    int f = f0 + col - 1;
    float v = 0.0f;
    if (t >= 0 && t < T_DIM && f >= 0 && f < F_DIM)
      v = x[((size_t)(b * C_DIM + ci) * T_DIM + t) * F_DIM + f];
    lds_x[r][col][ci] = (__bf16)v;
  }
  __syncthreads();

  v8f acc[4][2];
#pragma unroll
  for (int mt = 0; mt < 4; ++mt)
#pragma unroll
    for (int nt = 0; nt < 2; ++nt)
#pragma unroll
      for (int r = 0; r < 8; ++r) acc[mt][nt][r] = 0.0f;

  const int ka = half * 16;  // A K-offset pattern (contiguous, matches B)
  const int kb = half * 16;  // B K-offset pattern (contiguous, matches A)

#pragma unroll 1
  for (int tap = 0; tap < 9; ++tap) {
    const int kh = tap / 3, kw = tap % 3;
    if (tap < 8)  // prefetch next tap's weight panel into caches
      __builtin_prefetch(
          wconv + ((size_t)((tap + 1) * C_DIM + mgrp * 64 + ln) * C_DIM), 0, 1);
#pragma unroll 1
    for (int kc = 0; kc < 4; ++kc) {
      const int kbase = kc * 32;

      v16bf afrag[4];
#pragma unroll
      for (int mt = 0; mt < 4; ++mt) {
        const int co = (mgrp * 4 + mt) * 16 + ln;
        const __bf16* wp = wconv + ((size_t)(tap * C_DIM + co) * C_DIM) + kbase;
#pragma unroll
        for (int i = 0; i < 8; ++i) afrag[mt][i] = wp[ka + i];
#pragma unroll
        for (int i = 0; i < 8; ++i) afrag[mt][8 + i] = wp[ka + 8 + i];
      }

      v16bf bfrag[2];
#pragma unroll
      for (int nt = 0; nt < 2; ++nt) {
        const int n = ngrp * 2 + nt;
        const int tl = n >> 1;
        const int fl = ((n & 1) << 4) + ln;
        const __bf16* bp = &lds_x[tl + kh][fl + kw][kbase + kb];
#pragma unroll
        for (int i = 0; i < 16; ++i) bfrag[nt][i] = bp[i];
      }

#pragma unroll
      for (int mt = 0; mt < 4; ++mt)
#pragma unroll
        for (int nt = 0; nt < 2; ++nt)
          acc[mt][nt] = __builtin_amdgcn_wmma_f32_16x16x32_bf16(
              false, afrag[mt], false, bfrag[nt], (short)0, acc[mt][nt],
              false, false);
    }
  }

  // epilogue: bias, store h (bf16), per-pixel partial reductions
#pragma unroll
  for (int nt = 0; nt < 2; ++nt) {
    const int n = ngrp * 2 + nt;
    const int tl = n >> 1;
    const int fl = ((n & 1) << 4) + ln;
    const int t = t0 + tl, f = f0 + fl;
    float p1 = 0.0f, p2 = 0.0f;
#pragma unroll
    for (int mt = 0; mt < 4; ++mt) {
#pragma unroll
      for (int r = 0; r < 8; ++r) {
        const int com = (mgrp * 4 + mt) * 16 + half * 8 + r;  // D row (ISA C/D)
        float hv = acc[mt][nt][r] + lds_cb[com];
        h_out[(size_t)(b * C_DIM + com) * NPIX + t * F_DIM + f] = (__bf16)hv;
        p2 += hv * hv;
        p1 += hv * lds_sw[com];
      }
    }
    lds_p1[n * 16 + ln][mgrp * 2 + half] = p1;
    lds_p2[n * 16 + ln][mgrp * 2 + half] = p2;
  }
  __syncthreads();

  if (tid < TT * FT) {
    float s1 = 0.0f, s2 = 0.0f;
#pragma unroll
    for (int q = 0; q < 4; ++q) { s1 += lds_p1[tid][q]; s2 += lds_p2[tid][q]; }
    const int t = t0 + (tid >> 5);
    const int f = f0 + (tid & 31);
    logits[(size_t)b * NPIX + t * F_DIM + f] =
        s1 / (sqrtf(s2) + 1e-8f) + select_b[0];
  }
}

// ---------------------------------------------------------------------------
// 3) softmax per batch (in-place logits -> probs; also writes s_out)
// ---------------------------------------------------------------------------
__global__ __launch_bounds__(256) void softmax_kernel(
    float* __restrict__ probs, float* __restrict__ s_out) {
  __shared__ float red[256];
  const int b = blockIdx.x, tid = threadIdx.x;
  float* p = probs + (size_t)b * NPIX;

  float m = -1e30f;
  for (int i = tid; i < NPIX; i += 256) m = fmaxf(m, p[i]);
  red[tid] = m;
  __syncthreads();
  for (int s = 128; s > 0; s >>= 1) {
    if (tid < s) red[tid] = fmaxf(red[tid], red[tid + s]);
    __syncthreads();
  }
  m = red[0];
  __syncthreads();

  float sum = 0.0f;
  for (int i = tid; i < NPIX; i += 256) sum += __expf(p[i] - m);
  red[tid] = sum;
  __syncthreads();
  for (int s = 128; s > 0; s >>= 1) {
    if (tid < s) red[tid] += red[tid + s];
    __syncthreads();
  }
  const float inv = 1.0f / red[0];

  for (int i = tid; i < NPIX; i += 256) {
    float v = __expf(p[i] - m) * inv;
    p[i] = v;
    s_out[(size_t)b * NPIX + i] = v;
  }
}

// ---------------------------------------------------------------------------
// 4) top-k 512 per batch: radix-select threshold + compact + bitonic sort
// ---------------------------------------------------------------------------
__global__ __launch_bounds__(256) void topk_kernel(
    const float* __restrict__ probs, int* __restrict__ out_idx,
    int* __restrict__ sel_idx) {
  __shared__ unsigned hist[256];
  __shared__ unsigned long long skeys[K_SEL];
  __shared__ int lds_cnt;
  __shared__ unsigned s_prefix, s_mask;
  __shared__ int s_remaining;

  const int b = blockIdx.x, tid = threadIdx.x;
  const float* p = probs + (size_t)b * NPIX;

  if (tid == 0) { s_prefix = 0u; s_mask = 0u; s_remaining = K_SEL; lds_cnt = 0; }
  __syncthreads();

  // probs > 0, so uint bit pattern is order-preserving
  for (int pass = 0; pass < 4; ++pass) {
    const int shift = 24 - 8 * pass;
    hist[tid] = 0u;
    __syncthreads();
    const unsigned pref = s_prefix, msk = s_mask;
    for (int i = tid; i < NPIX; i += 256) {
      unsigned key = __float_as_uint(p[i]);
      if ((key & msk) == pref) atomicAdd(&hist[(key >> shift) & 255u], 1u);
    }
    __syncthreads();
    if (tid == 0) {
      int rem = s_remaining, cum = 0, bsel = 0;
      for (int bb = 255; bb >= 0; --bb) {
        int c = (int)hist[bb];
        if (cum + c >= rem) { bsel = bb; break; }
        cum += c;
      }
      s_prefix = pref | ((unsigned)bsel << shift);
      s_mask = msk | (0xFFu << shift);
      s_remaining = rem - cum;
    }
    __syncthreads();
  }
  const unsigned tkey = s_prefix;

  for (int i = tid; i < NPIX; i += 256) {
    unsigned key = __float_as_uint(p[i]);
    if (key > tkey) {
      int pos = atomicAdd(&lds_cnt, 1);
      if (pos < K_SEL)
        skeys[pos] = ((unsigned long long)key << 32) |
                     (unsigned)(0xFFFFFFFFu - (unsigned)i);
    }
  }
  __syncthreads();
  for (int i = tid; i < NPIX; i += 256) {
    unsigned key = __float_as_uint(p[i]);
    if (key == tkey) {
      int pos = atomicAdd(&lds_cnt, 1);
      if (pos < K_SEL)
        skeys[pos] = ((unsigned long long)key << 32) |
                     (unsigned)(0xFFFFFFFFu - (unsigned)i);
    }
  }
  __syncthreads();
  for (int i = lds_cnt + tid; i < K_SEL; i += 256) skeys[i] = 0ull;
  __syncthreads();

  // bitonic ascending sort of 512 packed keys
  for (int k2 = 2; k2 <= K_SEL; k2 <<= 1) {
    for (int j = k2 >> 1; j > 0; j >>= 1) {
      for (int i = tid; i < K_SEL; i += 256) {
        int ixj = i ^ j;
        if (ixj > i) {
          unsigned long long a = skeys[i], c = skeys[ixj];
          bool up = ((i & k2) == 0);
          if (up ? (a > c) : (a < c)) { skeys[i] = c; skeys[ixj] = a; }
        }
      }
      __syncthreads();
    }
  }

  for (int j = tid; j < K_SEL; j += 256) {
    unsigned long long e = skeys[K_SEL - 1 - j];  // largest first, ties: low idx
    int idx = (int)(0xFFFFFFFFu - (unsigned)(e & 0xFFFFFFFFu));
    out_idx[b * K_SEL + j] = idx;
    sel_idx[b * K_SEL + j] = idx;
  }
}

// ---------------------------------------------------------------------------
// 5) gather h at selected pixels + values GEMM (bf16 WMMA) -> latents
//    block = 8 waves: co 0..127 x 16 selected pixels
// ---------------------------------------------------------------------------
__global__ __launch_bounds__(256) void latents_kernel(
    const __bf16* __restrict__ h, const __bf16* __restrict__ wval,
    const float* __restrict__ values_b, const float* __restrict__ probs,
    const int* __restrict__ sel_idx, float* __restrict__ latents) {
  __shared__ __bf16 lds_h[16][CIP];
  __shared__ float lds_val[16];
  __shared__ int lds_idx[16];
  __shared__ float lds_vb[C_DIM];

  const int tid = threadIdx.x;
  const int lane = tid & 31, wv = tid >> 5;
  const int ln = lane & 15, half = lane >> 4;
  const int jt = blockIdx.x;  // pixel-tile 0..31
  const int b  = blockIdx.y;

  if (tid < 16) {
    int idx = sel_idx[b * K_SEL + jt * 16 + tid];
    lds_idx[tid] = idx;
    float sv = probs[(size_t)b * NPIX + idx];
    lds_val[tid] = sv + (1.0f - sv);
  }
  if (tid < C_DIM) lds_vb[tid] = values_b[tid];
  __syncthreads();

  for (int i = tid; i < 16 * C_DIM; i += 256) {
    int pl = i >> 7;
    int ci = i & 127;
    lds_h[pl][ci] = h[(size_t)(b * C_DIM + ci) * NPIX + lds_idx[pl]];
  }
  __syncthreads();

  v8f acc;
#pragma unroll
  for (int r = 0; r < 8; ++r) acc[r] = 0.0f;
  const int co = wv * 16 + ln;
  const int ka = half * 16, kb = half * 16;

#pragma unroll 1
  for (int kc = 0; kc < 4; ++kc) {
    const int kbase = kc * 32;
    const __bf16* wp = wval + (size_t)co * C_DIM + kbase;
    v16bf afrag;
#pragma unroll
    for (int i = 0; i < 8; ++i) afrag[i] = wp[ka + i];
#pragma unroll
    for (int i = 0; i < 8; ++i) afrag[8 + i] = wp[ka + 8 + i];
    const __bf16* bp = &lds_h[ln][kbase + kb];
    v16bf bfrag;
#pragma unroll
    for (int i = 0; i < 16; ++i) bfrag[i] = bp[i];
    acc = __builtin_amdgcn_wmma_f32_16x16x32_bf16(
        false, afrag, false, bfrag, (short)0, acc, false, false);
  }

  const float vmul = lds_val[ln];
#pragma unroll
  for (int r = 0; r < 8; ++r) {
    const int com = wv * 16 + half * 8 + r;
    const int j = jt * 16 + ln;
    latents[(size_t)(b * K_SEL + j) * C_DIM + com] =
        (acc[r] + lds_vb[com]) * vmul;
  }
}

// ---------------------------------------------------------------------------
extern "C" void kernel_launch(void* const* d_in, const int* in_sizes, int n_in,
                              void* d_out, int out_size, void* d_ws,
                              size_t ws_size, hipStream_t stream) {
  const float* x        = (const float*)d_in[0];
  const float* conv_w   = (const float*)d_in[1];
  const float* conv_b   = (const float*)d_in[2];
  const float* values_w = (const float*)d_in[3];
  const float* values_b = (const float*)d_in[4];
  const float* select_w = (const float*)d_in[5];
  const float* select_b = (const float*)d_in[6];

  char* ws = (char*)d_ws;
  __bf16* wconv = (__bf16*)(ws + WS_WCONV);
  __bf16* wval  = (__bf16*)(ws + WS_WVAL);
  __bf16* h     = (__bf16*)(ws + WS_H);
  float*  probs = (float*)(ws + WS_PROBS);
  int*    selid = (int*)(ws + WS_SEL);

  int*   out_idx  = (int*)d_out;                                   // B*K ints
  float* out_lat  = (float*)d_out + (B_DIM * K_SEL);               // B*K*C f32
  float* out_sout = (float*)d_out + (B_DIM * K_SEL) + (B_DIM * K_SEL * C_DIM);

  convert_w_kernel<<<(9 * 128 * 128 + 128 * 128 + 255) / 256, 256, 0, stream>>>(
      conv_w, values_w, wconv, wval);
  conv_select_kernel<<<dim3(F_DIM / FT, T_DIM / TT, B_DIM), 256, 0, stream>>>(
      x, wconv, conv_b, select_w, select_b, h, probs);
  softmax_kernel<<<B_DIM, 256, 0, stream>>>(probs, out_sout);
  topk_kernel<<<B_DIM, 256, 0, stream>>>(probs, out_idx, selid);
  latents_kernel<<<dim3(K_SEL / 16, B_DIM), 256, 0, stream>>>(
      h, wval, values_b, probs, selid, out_lat);
}
